// Net_12421045420310
// MI455X (gfx1250) — matrix operand order repaired and checked by reading, not compiled, not run on hardware
//
#include <hip/hip_runtime.h>
#include <hip/hip_bf16.h>
#include <stdint.h>

// ---------------------------------------------------------------------------
// Problem constants (from reference)
// ---------------------------------------------------------------------------
constexpr int N_NODES = 25000;
constexpr int N_EDGE  = 50000;
constexpr int N_EDGE3 = 40000;
constexpr int DIM     = 64;          // node feature dim
constexpr int H1      = 128;         // edge MLP hidden
constexpr int WCOLS   = DIM * DIM;   // 4096 per-edge weight entries
constexpr int MT      = N_EDGE / 16; // 3125 M-tiles (E divisible by 16)
constexpr int MG      = (MT + 3) / 4; // 782 groups of 4 M-tiles

typedef __attribute__((ext_vector_type(16))) __bf16 v16bf;
typedef __attribute__((ext_vector_type(8)))  float  v8f;

__device__ __forceinline__ float bf16lo(unsigned u) { return __uint_as_float(u << 16); }
__device__ __forceinline__ float bf16hi(unsigned u) { return __uint_as_float(u & 0xffff0000u); }

// ---------------------------------------------------------------------------
// Utility: zero a float buffer
// ---------------------------------------------------------------------------
__global__ __launch_bounds__(256) void zero_f32(float* __restrict__ p, int n) {
    int i = blockIdx.x * 256 + threadIdx.x;
    if (i < n) p[i] = 0.0f;
}

// ---------------------------------------------------------------------------
// Node encoder: h0 = relu(x @ W_node + b_node)   [N,8]x[8,64] -> [N,64]
// ---------------------------------------------------------------------------
__global__ __launch_bounds__(256) void node_enc(
    const float* __restrict__ x, const float* __restrict__ Wn,
    const float* __restrict__ bn, float* __restrict__ h0) {
    int id = blockIdx.x * 256 + threadIdx.x;
    if (id >= N_NODES * DIM) return;
    int n = id >> 6, d = id & 63;
    float acc = bn[d];
#pragma unroll
    for (int i = 0; i < 8; ++i) acc = fmaf(x[n * 8 + i], Wn[i * DIM + d], acc);
    h0[id] = fmaxf(acc, 0.0f);
}

// ---------------------------------------------------------------------------
// Edge attr layer: ea = relu(edge_attr @ W_ea + b_ea)   [E,19]x[19,12]
// ---------------------------------------------------------------------------
__global__ __launch_bounds__(256) void edge_ea(
    const float* __restrict__ eattr, const float* __restrict__ W,
    const float* __restrict__ b, float* __restrict__ ea) {
    int id = blockIdx.x * 256 + threadIdx.x;
    if (id >= N_EDGE * 12) return;
    int e = id / 12, j = id % 12;
    float acc = b[j];
#pragma unroll
    for (int i = 0; i < 19; ++i) acc = fmaf(eattr[e * 19 + i], W[i * 12 + j], acc);
    ea[id] = fmaxf(acc, 0.0f);
}

// ---------------------------------------------------------------------------
// h1 = relu(ea @ W_e1 + b_e1), written directly into the CDNA5 WMMA 16-bit
// A-matrix (16x32) lane layout, bf16, tiled [mt][kt][lane32][elem16].
//   lane<16  holds K in {0..7} U {16..23}; lane>=16 holds {8..15} U {24..31}
// ---------------------------------------------------------------------------
__global__ __launch_bounds__(256) void h1_pack(
    const float* __restrict__ ea, const float* __restrict__ W,
    const float* __restrict__ b, __hip_bfloat16* __restrict__ h1p) {
    int id = blockIdx.x * 256 + threadIdx.x;
    if (id >= N_EDGE * H1) return;
    int e = id >> 7, k = id & 127;
    float acc = b[k];
#pragma unroll
    for (int i = 0; i < 12; ++i) acc = fmaf(ea[e * 12 + i], W[i * H1 + k], acc);
    float v = fmaxf(acc, 0.0f);
    int mt = e >> 4, r = e & 15;
    int kt = k >> 5, kk = k & 31;
    int lane = r + ((kk & 8) ? 16 : 0);
    int elem = (kk < 16) ? (kk & 7) : ((kk & 7) + 8);
    size_t idx = (((size_t)mt * 4 + kt) * 32 + lane) * 16 + elem;
    h1p[idx] = __float2bfloat16(v);
}

// ---------------------------------------------------------------------------
// Repack W_e2 [128,4096] f32 into WMMA 16-bit B-matrix (32x16) layout, bf16,
// tiled [nt][kt][lane32][elem16]: lanes 0-15 hold K {0..7,16..23}, lanes 16-31 K {8..15,24..31}
// ---------------------------------------------------------------------------
__global__ __launch_bounds__(256) void repack_we2(
    const float* __restrict__ We2, __hip_bfloat16* __restrict__ we2p) {
    int id = blockIdx.x * 256 + threadIdx.x;
    if (id >= H1 * WCOLS) return;
    int k = id / WCOLS, c = id % WCOLS;
    int nt = c >> 4, n = c & 15;
    int kt = k >> 5, kk = k & 31;
    int lane = n + ((kk & 8) ? 16 : 0);
    int elem = (kk < 16) ? (kk & 7) : ((kk & 7) + 8);
    size_t idx = (((size_t)nt * 4 + kt) * 32 + lane) * 16 + elem;
    we2p[idx] = __float2bfloat16(We2[id]);
}

// ---------------------------------------------------------------------------
// Big GEMM: Wedge[e, c] = h1[e,:] @ W_e2[:, c] + b_e2[c]  (bf16 out)
// Block = 4 waves sharing one 64-column B strip (16KB) staged into LDS via
// CDNA5 async-to-LDS copies; each wave owns one 16-row M-tile with all 4 A
// K-tiles held in VGPRs. Inner loop: ds_load B + v_wmma_f32_16x16x32_bf16.
// D tiles staged through LDS so global stores are coalesced b128 stores.
// ---------------------------------------------------------------------------
__global__ __launch_bounds__(128) void gemm_wedge(
    const __hip_bfloat16* __restrict__ h1p_,
    const __hip_bfloat16* __restrict__ we2p_,
    const float* __restrict__ b_e2,
    __hip_bfloat16* __restrict__ wedge) {
    __shared__ __align__(16) __hip_bfloat16 Bsh[4 * 4 * 32 * 16];   // 16KB B strip
    __shared__ __align__(16) __hip_bfloat16 stage[4][16][64];       // 8KB D stage
    const int wv   = threadIdx.x >> 5;
    const int lane = threadIdx.x & 31;
    const int mg = blockIdx.x >> 6;        // group of 4 M-tiles
    const int ns = blockIdx.x & 63;        // 64-column strip index
    const int mt = mg * 4 + wv;
    const bool valid = (mt < MT);
    const int mtc = valid ? mt : (MT - 1);

    // --- async copy of this block's 16KB B strip into LDS (ASYNCcnt path) ---
    {
        unsigned lds0 = (unsigned)(uintptr_t)(&Bsh[0]) + (unsigned)threadIdx.x * 16u;
        unsigned long long g0 = (unsigned long long)(uintptr_t)we2p_ +
                                (unsigned long long)ns * 16384ull +
                                (unsigned long long)threadIdx.x * 16ull;
#pragma unroll
        for (int r = 0; r < 8; ++r) {
            asm volatile("global_load_async_to_lds_b128 %0, %1, off"
                         :: "v"(lds0 + (unsigned)(r * 2048)),
                            "v"(g0 + (unsigned long long)(r * 2048))
                         : "memory");
        }
        asm volatile("s_wait_asynccnt 0x0" ::: "memory");
    }

    // --- A K-tiles into registers (global, contiguous 32B/lane) ---
    const v16bf* A = (const v16bf*)h1p_;
    v16bf a[4];
#pragma unroll
    for (int kt = 0; kt < 4; ++kt) a[kt] = A[((size_t)mtc * 4 + kt) * 32 + lane];

    __syncthreads();   // B strip visible to all waves

    // --- MMA loop: B from LDS, A from registers ---
    const v16bf* B = (const v16bf*)Bsh;
    v8f acc[4] = {};
#pragma unroll
    for (int nt = 0; nt < 4; ++nt) {
#pragma unroll
        for (int kt = 0; kt < 4; ++kt) {
            v16bf bm = B[(nt * 4 + kt) * 32 + lane];
            acc[nt] = __builtin_amdgcn_wmma_f32_16x16x32_bf16(
                false, a[kt], false, bm, (short)0, acc[nt], false, false);
        }
    }

    // D layout: lane<16 -> rows 0..7 in vgpr v, lane>=16 -> rows 8..15; col = lane&15
    const int nlo = lane & 15;
    const int mbase = (lane >> 4) ? 8 : 0;
#pragma unroll
    for (int nt = 0; nt < 4; ++nt) {
        int col = nt * 16 + nlo;
        float bias = b_e2[ns * 64 + col];
#pragma unroll
        for (int v = 0; v < 8; ++v) {
            stage[wv][mbase + v][col] = __float2bfloat16(acc[nt][v] + bias);
        }
    }
    __syncthreads();

    // Coalesced store: 4 rounds; each round a 32-lane b128 store covers 4 rows
    if (valid) {
        const size_t ebase = (size_t)mt * 16;
#pragma unroll
        for (int r = 0; r < 4; ++r) {
            int row = r * 4 + (lane >> 3);
            int cb  = (lane & 7) * 8;   // bf16 offset within row (16B granules)
            const float4* s = (const float4*)&stage[wv][row][cb];
            float4* dptr = (float4*)(wedge + (ebase + row) * (size_t)WCOLS + ns * 64 + cb);
            *dptr = *s;
        }
    }
}

// ---------------------------------------------------------------------------
// In-degree count (float, for scatter-mean)
// ---------------------------------------------------------------------------
__global__ __launch_bounds__(256) void count_deg(const int* __restrict__ ei,
                                                 float* __restrict__ cnt) {
    int e = blockIdx.x * 256 + threadIdx.x;
    if (e < N_EDGE) atomicAdd(&cnt[ei[N_EDGE + e]], 1.0f);
}

// ---------------------------------------------------------------------------
// Per-edge GEMV + scatter-add:
//   msg[e,f] = sum_d h[src[e],d] * Wedge[e, d*64+f];  agg[dst[e],f] += msg
// 32 threads per edge, each handles a bf16 pair of f; source row cached in LDS.
// ---------------------------------------------------------------------------
__global__ __launch_bounds__(256) void conv_msg(
    const float* __restrict__ h, const __hip_bfloat16* __restrict__ wedge,
    const int* __restrict__ ei, float* __restrict__ agg) {
    __shared__ float sh[8][64];
    int t = threadIdx.x;
    int le = t >> 5;        // edge slot in block (8 edges/block)
    int lf = t & 31;        // lane -> f pair index
    size_t e = (size_t)blockIdx.x * 8 + le;
    int src = ei[e];
    int dst = ei[N_EDGE + e];
    sh[le][lf]      = h[(size_t)src * 64 + lf];
    sh[le][lf + 32] = h[(size_t)src * 64 + lf + 32];

    const unsigned* w = (const unsigned*)(wedge + e * (size_t)WCOLS) + lf;
    __builtin_prefetch((const void*)(w + 1024), 0, 1);   // second half of row
    __syncthreads();

    float a0 = 0.0f, a1 = 0.0f;
#pragma unroll 8
    for (int d = 0; d < 64; ++d) {
        unsigned u = w[d * 32];
        float s = sh[le][d];
        a0 = fmaf(s, bf16lo(u), a0);
        a1 = fmaf(s, bf16hi(u), a1);
    }
    atomicAdd(&agg[(size_t)dst * 64 + 2 * lf],     a0);
    atomicAdd(&agg[(size_t)dst * 64 + 2 * lf + 1], a1);
}

// ---------------------------------------------------------------------------
// m = relu(agg / max(cnt,1) + conv_bias)
// ---------------------------------------------------------------------------
__global__ __launch_bounds__(256) void relu_mean(
    const float* __restrict__ agg, const float* __restrict__ cnt,
    const float* __restrict__ cb, float* __restrict__ m) {
    int id = blockIdx.x * 256 + threadIdx.x;
    if (id >= N_NODES * DIM) return;
    int n = id >> 6, d = id & 63;
    float v = agg[id] / fmaxf(cnt[n], 1.0f) + cb[d];
    m[id] = fmaxf(v, 0.0f);
}

// ---------------------------------------------------------------------------
// Fused single-step GRU (PyTorch gate order r,z,n). One thread per (node,f).
// ---------------------------------------------------------------------------
__global__ __launch_bounds__(256) void gru_step(
    const float* __restrict__ m, const float* __restrict__ hin,
    const float* __restrict__ Wih, const float* __restrict__ bih,
    const float* __restrict__ Whh, const float* __restrict__ bhh,
    float* __restrict__ hout) {
    int id = blockIdx.x * 256 + threadIdx.x;
    if (id >= N_NODES * DIM) return;
    int n = id >> 6, f = id & 63;
    float ir = bih[f], iz = bih[64 + f], in_ = bih[128 + f];
    float hr = bhh[f], hz = bhh[64 + f], hn  = bhh[128 + f];
    const float* mr = m   + (size_t)n * 64;
    const float* hrow = hin + (size_t)n * 64;
#pragma unroll 4
    for (int d = 0; d < 64; ++d) {
        float mv = mr[d], hv = hrow[d];
        ir  = fmaf(mv, Wih[f * 64 + d],        ir);
        iz  = fmaf(mv, Wih[(64 + f) * 64 + d], iz);
        in_ = fmaf(mv, Wih[(128 + f) * 64 + d], in_);
        hr  = fmaf(hv, Whh[f * 64 + d],        hr);
        hz  = fmaf(hv, Whh[(64 + f) * 64 + d], hz);
        hn  = fmaf(hv, Whh[(128 + f) * 64 + d], hn);
    }
    float r = 1.0f / (1.0f + expf(-(ir + hr)));
    float z = 1.0f / (1.0f + expf(-(iz + hz)));
    float ng = tanhf(in_ + r * hn);
    hout[id] = (1.0f - z) * ng + z * hrow[f];
}

// ---------------------------------------------------------------------------
// Final readout: one block (128 threads) per e3 edge.
// pair = 0.5*(h[i0]+h[i1]); yhat = relu([pair,ea3] @ W_l1 + b_l1) @ W_l2 + b_l2
// ---------------------------------------------------------------------------
__global__ __launch_bounds__(128) void final_mlp(
    const float* __restrict__ h, const float* __restrict__ ea3,
    const int* __restrict__ ei3,
    const float* __restrict__ Wl1, const float* __restrict__ bl1,
    const float* __restrict__ Wl2, const float* __restrict__ bl2,
    float* __restrict__ out) {
    __shared__ float pv[72];
    __shared__ float red[128];
    int e3 = blockIdx.x, t = threadIdx.x;
    int i0 = ei3[e3], i1 = ei3[N_EDGE3 + e3];
    if (t < 64)      pv[t] = 0.5f * (h[(size_t)i0 * 64 + t] + h[(size_t)i1 * 64 + t]);
    else if (t < 72) pv[t] = ea3[(size_t)e3 * 8 + (t - 64)];
    __syncthreads();
    float acc = bl1[t];
#pragma unroll 8
    for (int d = 0; d < 72; ++d) acc = fmaf(pv[d], Wl1[d * 128 + t], acc);
    float hid = fmaxf(acc, 0.0f);
    red[t] = hid * Wl2[t];
    __syncthreads();
#pragma unroll
    for (int s = 64; s > 0; s >>= 1) {
        if (t < s) red[t] += red[t + s];
        __syncthreads();
    }
    if (t == 0) out[e3] = red[0] + bl2[0];
}

// ---------------------------------------------------------------------------
// Host launcher
// ---------------------------------------------------------------------------
extern "C" void kernel_launch(void* const* d_in, const int* in_sizes, int n_in,
                              void* d_out, int out_size, void* d_ws, size_t ws_size,
                              hipStream_t stream) {
    (void)in_sizes; (void)n_in; (void)out_size; (void)ws_size;

    const float* x      = (const float*)d_in[0];
    const float* eattr  = (const float*)d_in[1];
    const float* eattr3 = (const float*)d_in[2];
    const int*   ei     = (const int*)  d_in[3];
    const int*   ei3    = (const int*)  d_in[4];
    const float* W_node = (const float*)d_in[5];
    const float* b_node = (const float*)d_in[6];
    const float* W_ea   = (const float*)d_in[7];
    const float* b_ea   = (const float*)d_in[8];
    const float* W_e1   = (const float*)d_in[9];
    const float* b_e1   = (const float*)d_in[10];
    const float* W_e2   = (const float*)d_in[11];
    const float* b_e2   = (const float*)d_in[12];
    const float* conv_b = (const float*)d_in[13];
    const float* W_ih   = (const float*)d_in[14];
    const float* b_ih   = (const float*)d_in[15];
    const float* W_hh   = (const float*)d_in[16];
    const float* b_hh   = (const float*)d_in[17];
    const float* W_l1   = (const float*)d_in[18];
    const float* b_l1   = (const float*)d_in[19];
    const float* W_l2   = (const float*)d_in[20];
    const float* b_l2   = (const float*)d_in[21];

    // Workspace carve-out (256B aligned)
    char* ws = (char*)d_ws;
    size_t off = 0;
    auto take = [&](size_t bytes) -> char* {
        char* p = ws + off;
        off = (off + bytes + 255) & ~(size_t)255;
        return p;
    };
    float* h0   = (float*)take((size_t)N_NODES * DIM * 4);
    float* h1   = (float*)take((size_t)N_NODES * DIM * 4);
    float* ea   = (float*)take((size_t)N_EDGE * 12 * 4);
    __hip_bfloat16* h1p   = (__hip_bfloat16*)take((size_t)N_EDGE * H1 * 2);
    __hip_bfloat16* we2p  = (__hip_bfloat16*)take((size_t)H1 * WCOLS * 2);
    __hip_bfloat16* wedge = (__hip_bfloat16*)take((size_t)N_EDGE * WCOLS * 2);
    float* cnt  = (float*)take((size_t)N_NODES * 4);
    float* agg  = (float*)take((size_t)N_NODES * DIM * 4);
    float* mbuf = (float*)take((size_t)N_NODES * DIM * 4);

    // --- Stage 1: encoders + operand packing
    node_enc<<<(N_NODES * DIM) / 256, 256, 0, stream>>>(x, W_node, b_node, h0);
    edge_ea<<<(N_EDGE * 12 + 255) / 256, 256, 0, stream>>>(eattr, W_ea, b_ea, ea);
    h1_pack<<<(N_EDGE * H1) / 256, 256, 0, stream>>>(ea, W_e1, b_e1, h1p);
    repack_we2<<<(H1 * WCOLS) / 256, 256, 0, stream>>>(W_e2, we2p);

    // --- Stage 2: the big WMMA GEMM producing per-edge weight matrices
    gemm_wedge<<<MG * 64, 128, 0, stream>>>(h1p, we2p, b_e2, wedge);

    // --- Stage 3: degree counts
    zero_f32<<<(N_NODES + 255) / 256, 256, 0, stream>>>(cnt, N_NODES);
    count_deg<<<(N_EDGE + 255) / 256, 256, 0, stream>>>(ei, cnt);

    // --- Stage 4: 3 rounds of conv + GRU (ping-pong h0/h1)
    for (int it = 0; it < 3; ++it) {
        float* hin  = (it & 1) ? h1 : h0;
        float* hout = (it & 1) ? h0 : h1;
        zero_f32<<<(N_NODES * DIM) / 256, 256, 0, stream>>>(agg, N_NODES * DIM);
        conv_msg<<<N_EDGE / 8, 256, 0, stream>>>(hin, wedge, ei, agg);
        relu_mean<<<(N_NODES * DIM) / 256, 256, 0, stream>>>(agg, cnt, conv_b, mbuf);
        gru_step<<<(N_NODES * DIM) / 256, 256, 0, stream>>>(mbuf, hin, W_ih, b_ih,
                                                            W_hh, b_hh, hout);
    }

    // --- Stage 5: readout (after 3 iters the live buffer is h1)
    final_mlp<<<N_EDGE3, 128, 0, stream>>>(h1, eattr3, ei3, W_l1, b_l1, W_l2, b_l2,
                                           (float*)d_out);
}
